// EX_Module_51110110822908
// MI455X (gfx1250) — matrix operand order repaired and checked by reading, compiled, not run on hardware
//
#include <hip/hip_runtime.h>
#include <hip/hip_bf16.h>

#define BATCH 16
#define CIN   512
#define CMID  256
#define ATTN  32
#define HW    4096   // 64*64

typedef _Float16 v16h __attribute__((ext_vector_type(16)));
typedef _Float16 v8h  __attribute__((ext_vector_type(8)));
typedef _Float16 v4h  __attribute__((ext_vector_type(4)));
typedef float    v8f  __attribute__((ext_vector_type(8)));

#define SHUF16(lo, hi) __builtin_shufflevector((lo), (hi), 0,1,2,3,4,5,6,7,8,9,10,11,12,13,14,15)

// ---------------------------------------------------------------------------
// Kernel: r[b,h] = sum_c w[b?]*x[b,c,h] + bias   (dot over channels, per pixel)
// float2 per thread; used for q (w_bstride=0) and ctx (per-b weights).
// ---------------------------------------------------------------------------
__global__ __launch_bounds__(256) void dot_over_c_kernel(
    const float* __restrict__ x, const float* __restrict__ w, int w_bstride,
    const float* __restrict__ bias, int bias_stride, float* __restrict__ out)
{
    const int b = blockIdx.x / (HW / 512);
    const int h = (blockIdx.x % (HW / 512)) * 512 + threadIdx.x * 2;
    __shared__ float wsm[CIN];
    const float* wb = w + (size_t)b * w_bstride;
    for (int i = threadIdx.x; i < CIN; i += 256) wsm[i] = wb[i];
    __syncthreads();
    const float* xb = x + ((size_t)b * CIN) * HW + h;
    float a0 = 0.f, a1 = 0.f;
    for (int c = 0; c < CIN; ++c) {
        const float wv = wsm[c];
        const float2 xv = *(const float2*)&xb[(size_t)c * HW];
        a0 = fmaf(wv, xv.x, a0);
        a1 = fmaf(wv, xv.y, a1);
    }
    const float bv = bias[b * bias_stride];
    float2 o; o.x = a0 + bv; o.y = a1 + bv;
    *(float2*)&out[(size_t)b * HW + h] = o;
}

// ---------------------------------------------------------------------------
// Kernel: row-wise softmax, one block per batch row of length N
// ---------------------------------------------------------------------------
__global__ __launch_bounds__(256) void softmax_kernel(
    const float* __restrict__ in, float* __restrict__ out, int N)
{
    const int b = blockIdx.x;
    const float* src = in + (size_t)b * N;
    float* dst = out + (size_t)b * N;
    __shared__ float red[256];
    float m = -3.4e38f;
    for (int i = threadIdx.x; i < N; i += 256) m = fmaxf(m, src[i]);
    red[threadIdx.x] = m; __syncthreads();
    for (int s = 128; s > 0; s >>= 1) {
        if (threadIdx.x < s) red[threadIdx.x] = fmaxf(red[threadIdx.x], red[threadIdx.x + s]);
        __syncthreads();
    }
    m = red[0]; __syncthreads();
    float sum = 0.f;
    for (int i = threadIdx.x; i < N; i += 256) sum += __expf(src[i] - m);
    red[threadIdx.x] = sum; __syncthreads();
    for (int s = 128; s > 0; s >>= 1) {
        if (threadIdx.x < s) red[threadIdx.x] += red[threadIdx.x + s];
        __syncthreads();
    }
    const float inv = 1.f / red[0];
    for (int i = threadIdx.x; i < N; i += 256) dst[i] = __expf(src[i] - m) * inv;
}

// ---------------------------------------------------------------------------
// Kernel: xm[b,c] = sum_h x[b,c,h] * mask[b,h]     (one block per (b,c))
// ---------------------------------------------------------------------------
__global__ __launch_bounds__(256) void dot_over_h_kernel(
    const float* __restrict__ x, const float* __restrict__ mask,
    float* __restrict__ xm)
{
    const int b = blockIdx.x / CIN;
    const int c = blockIdx.x % CIN;
    const float* xr = x + ((size_t)b * CIN + c) * HW;
    const float* mr = mask + (size_t)b * HW;
    float acc = 0.f;
    for (int i = threadIdx.x; i < HW; i += 256) acc = fmaf(xr[i], mr[i], acc);
    __shared__ float red[256];
    red[threadIdx.x] = acc; __syncthreads();
    for (int s = 128; s > 0; s >>= 1) {
        if (threadIdx.x < s) red[threadIdx.x] += red[threadIdx.x + s];
        __syncthreads();
    }
    if (threadIdx.x == 0) xm[blockIdx.x] = red[0];
}

// ---------------------------------------------------------------------------
// WMMA kernel: rowmax[b,m] = max_h ( w_ql[m,:] . x[b,:,h] ) + b_ql[m]
// grid.x = BATCH*16 (m-tiles of 16 rows), block = 256 threads (8 waves).
// Both operands staged in LDS in *fragment order* so every lane's 16 f16 are
// contiguous -> fragment build = 2x ds_load_b128. Each wave runs 2 h-tiles
// per A-fragment (2 WMMAs / 6 LDS b128 loads). x slab transposed into LDS
// with packed ds_store_b64 writes. Row-max folded in registers.
// ---------------------------------------------------------------------------
__global__ __launch_bounds__(256) void gemm_rowmax_kernel(
    const float* __restrict__ x, const float* __restrict__ w_ql,
    const float* __restrict__ b_ql, float* __restrict__ rowmax)
{
    const int b      = blockIdx.x >> 4;
    const int m_base = (blockIdx.x & 15) * 16;
    const int tid    = threadIdx.x;
    const int wave   = tid >> 5;
    const int lane   = tid & 31;
    const int half   = lane >> 4;   // K-half selector per ISA fragment layout
    const int col    = lane & 15;   // A: M row ; B: N column

    // A operand pre-expanded to per-lane fragment order: [kstep][lane][16(+8 pad)]
    __shared__ __align__(16) _Float16 Af[16][32][24];
    // x slab transposed: Xt[h_local][k_local], pitch 40 f16 (80B, 16B-aligned rows)
    __shared__ __align__(16) _Float16 Xt[256][40];
    __shared__ float wredS[8][16];

    // ---- one-time: expand 16x512 A tile into fragment-ordered LDS ----
    for (int i = tid; i < 16 * 32 * 16; i += 256) {
        const int e  = i & 15;
        const int ln = (i >> 4) & 31;
        const int ks = i >> 9;
        const int cc = ln & 15, hf = ln >> 4;
        const int j = e >> 1, p = e & 1;
        const int kl = 16 * hf + 2 * j + p;
        Af[ks][ln][e] = (_Float16)w_ql[(size_t)(m_base + cc) * CIN + ks * 32 + kl];
    }

    float rmax[8];
#pragma unroll
    for (int v = 0; v < 8; ++v) rmax[v] = -3.4e38f;

    const float* xb = x + (size_t)b * CIN * HW;

    for (int hc = 0; hc < HW / 256; ++hc) {
        const int h_base = hc * 256;
        v8f acc0 = {}, acc1 = {};
        for (int ks = 0; ks < 16; ++ks) {
            const int k0 = ks * 32;
            __syncthreads();
            // ---- stage 32(K) x 256(H) slab, transposed & packed ----
#pragma unroll
            for (int u0 = 0; u0 < 2; ++u0) {
                const int u   = u0 * 256 + tid;
                const int kk0 = (u >> 6) * 4;
                const int hh0 = (u & 63) * 4;
                const float* gsrc = &xb[(size_t)(k0 + kk0) * HW + h_base + hh0];
                const float4 r0 = *(const float4*)(gsrc);
                const float4 r1 = *(const float4*)(gsrc + HW);
                const float4 r2 = *(const float4*)(gsrc + 2 * HW);
                const float4 r3 = *(const float4*)(gsrc + 3 * HW);
                const float c0[4] = {r0.x, r0.y, r0.z, r0.w};
                const float c1[4] = {r1.x, r1.y, r1.z, r1.w};
                const float c2[4] = {r2.x, r2.y, r2.z, r2.w};
                const float c3[4] = {r3.x, r3.y, r3.z, r3.w};
#pragma unroll
                for (int c = 0; c < 4; ++c) {
                    v4h pk = {(_Float16)c0[c], (_Float16)c1[c],
                              (_Float16)c2[c], (_Float16)c3[c]};
                    *(v4h*)&Xt[hh0 + c][kk0] = pk;   // ds_store_b64
                }
            }
            __syncthreads();
            // ---- fragments: contiguous 16-byte LDS loads ----
            const v8h* ap = (const v8h*)&Af[ks][lane][0];
            const v16h a = SHUF16(ap[0], ap[1]);
            const v8h* b0p = (const v8h*)&Xt[wave * 16 + col][16 * half];
            const v16h bb0 = SHUF16(b0p[0], b0p[1]);
            const v8h* b1p = (const v8h*)&Xt[128 + wave * 16 + col][16 * half];
            const v16h bb1 = SHUF16(b1p[0], b1p[1]);
            acc0 = __builtin_amdgcn_wmma_f32_16x16x32_f16(
                false, a, false, bb0, (short)0, acc0, false, false);
            acc1 = __builtin_amdgcn_wmma_f32_16x16x32_f16(
                false, a, false, bb1, (short)0, acc1, false, false);
        }
#pragma unroll
        for (int v = 0; v < 8; ++v)
            rmax[v] = fmaxf(rmax[v], fmaxf(acc0[v], acc1[v]));
    }

    // cross-lane max within each 16-lane half (columns of the tile)
#pragma unroll
    for (int v = 0; v < 8; ++v) {
        float m = rmax[v];
        m = fmaxf(m, __shfl_xor(m, 1));
        m = fmaxf(m, __shfl_xor(m, 2));
        m = fmaxf(m, __shfl_xor(m, 4));
        m = fmaxf(m, __shfl_xor(m, 8));
        rmax[v] = m;   // row max for M = v + 8*half over this wave's columns
    }
    if (col == 0) {
#pragma unroll
        for (int v = 0; v < 8; ++v) wredS[wave][v + 8 * half] = rmax[v];
    }
    __syncthreads();
    if (tid < 16) {
        float m = wredS[0][tid];
#pragma unroll
        for (int w = 1; w < 8; ++w) m = fmaxf(m, wredS[w][tid]);
        rowmax[(size_t)b * CMID + m_base + tid] = m + b_ql[m_base + tid];
    }
}

// ---------------------------------------------------------------------------
// Kernel: v[b,c] = sum_m avg[b,m]*w_vl[m,c] ; cb[b] = avg[b,:].b_vl
// ---------------------------------------------------------------------------
__global__ __launch_bounds__(256) void vproj_kernel(
    const float* __restrict__ avg, const float* __restrict__ w_vl,
    const float* __restrict__ b_vl, float* __restrict__ vout,
    float* __restrict__ cb)
{
    const int b = blockIdx.x;
    __shared__ float av[CMID];
    __shared__ float red[256];
    av[threadIdx.x]  = avg[(size_t)b * CMID + threadIdx.x];
    red[threadIdx.x] = av[threadIdx.x] * b_vl[threadIdx.x];
    __syncthreads();
    for (int s = 128; s > 0; s >>= 1) {
        if (threadIdx.x < s) red[threadIdx.x] += red[threadIdx.x + s];
        __syncthreads();
    }
    if (threadIdx.x == 0) cb[b] = red[0];
    for (int c0 = 0; c0 < CIN; c0 += 256) {
        const int c = c0 + threadIdx.x;
        float acc = 0.f;
        for (int m = 0; m < CMID; ++m) acc = fmaf(av[m], w_vl[(size_t)m * CIN + c], acc);
        vout[(size_t)b * CIN + c] = acc;
    }
}

// ---------------------------------------------------------------------------
// Per-batch fused head: context -> up -> LayerNorm -> sigmoid (= ch), then
// s = ch*(1+1/HW)+1/HW (analytic collapse of stack mean), BN+relu reduce,
// select + 2-way softmax  ->  alpha[b,c], beta[b,c].
// ---------------------------------------------------------------------------
__global__ __launch_bounds__(512) void fuse_kernel(
    const float* __restrict__ xm,
    const float* __restrict__ w_vr, const float* __restrict__ b_vr,
    const float* __restrict__ w_up, const float* __restrict__ b_up,
    const float* __restrict__ ln_g, const float* __restrict__ ln_b,
    const float* __restrict__ w_red,
    const float* __restrict__ bn_g, const float* __restrict__ bn_b,
    const float* __restrict__ bn_rm, const float* __restrict__ bn_rv,
    const float* __restrict__ w_sel,
    float* __restrict__ alpha, float* __restrict__ beta)
{
    const int b = blockIdx.x;
    const int t = threadIdx.x;
    __shared__ float xs[CIN];
    __shared__ float cx[CMID];
    __shared__ float up[CIN];
    __shared__ float rd[ATTN];
    __shared__ float sred[512];

    xs[t] = xm[(size_t)b * CIN + t];
    __syncthreads();
    if (t < CMID) {
        float acc = b_vr[t];
        for (int c = 0; c < CIN; ++c) acc = fmaf(w_vr[(size_t)t * CIN + c], xs[c], acc);
        cx[t] = acc;
    }
    __syncthreads();
    {
        float acc = b_up[t];
        for (int m = 0; m < CMID; ++m) acc = fmaf(w_up[(size_t)t * CMID + m], cx[m], acc);
        up[t] = acc;
    }
    __syncthreads();
    sred[t] = up[t]; __syncthreads();
    for (int s = 256; s > 0; s >>= 1) {
        if (t < s) sred[t] += sred[t + s];
        __syncthreads();
    }
    const float mean = sred[0] * (1.f / CIN);
    __syncthreads();
    const float d = up[t] - mean;
    sred[t] = d * d; __syncthreads();
    for (int s = 256; s > 0; s >>= 1) {
        if (t < s) sred[t] += sred[t + s];
        __syncthreads();
    }
    const float rstd = rsqrtf(sred[0] * (1.f / CIN) + 1e-5f);
    const float ln = d * rstd * ln_g[t] + ln_b[t];
    const float ch = 1.f / (1.f + __expf(-ln));

    const float kinv = 1.f / (float)HW;
    __syncthreads();
    sred[t] = ch * (1.f + kinv) + kinv;    // s[b,c] analytic
    __syncthreads();
    if (t < ATTN) {
        float acc = 0.f;
        for (int c = 0; c < CIN; ++c) acc = fmaf(w_red[(size_t)t * CIN + c], sred[c], acc);
        acc = (acc - bn_rm[t]) * rsqrtf(bn_rv[t] + 1e-5f) * bn_g[t] + bn_b[t];
        rd[t] = fmaxf(acc, 0.f);
    }
    __syncthreads();
    float s0 = 0.f, s1 = 0.f;
    for (int a = 0; a < ATTN; ++a) {
        s0 = fmaf(w_sel[(size_t)t * ATTN + a], rd[a], s0);
        s1 = fmaf(w_sel[(size_t)(CIN + t) * ATTN + a], rd[a], s1);
    }
    const float mx = fmaxf(s0, s1);
    const float e0 = __expf(s0 - mx), e1 = __expf(s1 - mx);
    const float a0 = e0 / (e0 + e1), a1 = e1 / (e0 + e1);
    alpha[(size_t)b * CIN + t] = a0 * ch + a1;
    beta [(size_t)b * CIN + t] = a1 * ch;
}

// ---------------------------------------------------------------------------
// Epilogue: out = x * (1 + alpha[b,c]*sp[b,h] + beta[b,c])   (float4)
// ---------------------------------------------------------------------------
__global__ __launch_bounds__(256) void final_kernel(
    const float* __restrict__ x, const float* __restrict__ sp,
    const float* __restrict__ alpha, const float* __restrict__ beta,
    float* __restrict__ out)
{
    const size_t i = ((size_t)blockIdx.x * 256 + threadIdx.x) * 4;
    const int b = (int)(i / ((size_t)CIN * HW));
    const int c = (int)((i / HW) % CIN);
    const int h = (int)(i % HW);
    const float4 xv = *(const float4*)(x + i);
    const float4 sv = *(const float4*)(sp + (size_t)b * HW + h);
    const float al = alpha[(size_t)b * CIN + c];
    const float be = beta[(size_t)b * CIN + c];
    float4 o;
    o.x = xv.x * (1.f + al * sv.x + be);
    o.y = xv.y * (1.f + al * sv.y + be);
    o.z = xv.z * (1.f + al * sv.z + be);
    o.w = xv.w * (1.f + al * sv.w + be);
    *(float4*)(out + i) = o;
}

extern "C" void kernel_launch(void* const* d_in, const int* in_sizes, int n_in,
                              void* d_out, int out_size, void* d_ws, size_t ws_size,
                              hipStream_t stream) {
    const float* x    = (const float*)d_in[0];
    const float* w_qr = (const float*)d_in[1];
    const float* b_qr = (const float*)d_in[2];
    const float* w_vr = (const float*)d_in[3];
    const float* b_vr = (const float*)d_in[4];
    const float* w_up = (const float*)d_in[5];
    const float* b_up = (const float*)d_in[6];
    const float* ln_g = (const float*)d_in[7];
    const float* ln_b = (const float*)d_in[8];
    const float* w_ql = (const float*)d_in[9];
    const float* b_ql = (const float*)d_in[10];
    const float* w_vl = (const float*)d_in[11];
    const float* b_vl = (const float*)d_in[12];
    const float* w_red= (const float*)d_in[13];
    const float* bn_g = (const float*)d_in[14];
    const float* bn_b = (const float*)d_in[15];
    const float* bn_rm= (const float*)d_in[16];
    const float* bn_rv= (const float*)d_in[17];
    const float* w_sel= (const float*)d_in[18];
    float* out = (float*)d_out;

    float* ws      = (float*)d_ws;
    float* q       = ws;                 // [B,HW]
    float* mask    = ws + 65536;         // [B,HW]
    float* ctx     = ws + 131072;        // [B,HW]
    float* sp      = ws + 196608;        // [B,HW]
    float* xm      = ws + 262144;        // [B,CIN]
    float* rowmax_ = ws + 270336;        // [B,CMID]
    float* avg     = ws + 274432;        // [B,CMID]
    float* vp      = ws + 278528;        // [B,CIN]
    float* cb      = ws + 286720;        // [B]
    float* alpha   = ws + 286736;        // [B,CIN]
    float* beta    = ws + 294928;        // [B,CIN]

    // channel-attn path (GEMM folded away)
    dot_over_c_kernel<<<BATCH * (HW / 512), 256, 0, stream>>>(x, w_qr, 0, b_qr, 0, q);
    softmax_kernel<<<BATCH, 256, 0, stream>>>(q, mask, HW);
    dot_over_h_kernel<<<BATCH * CIN, 256, 0, stream>>>(x, mask, xm);

    // spatial path: the one real GEMM (WMMA, row-max reduced)
    gemm_rowmax_kernel<<<BATCH * 16, 256, 0, stream>>>(x, w_ql, b_ql, rowmax_);
    softmax_kernel<<<BATCH, 256, 0, stream>>>(rowmax_, avg, CMID);
    vproj_kernel<<<BATCH, 256, 0, stream>>>(avg, w_vl, b_vl, vp, cb);
    dot_over_c_kernel<<<BATCH * (HW / 512), 256, 0, stream>>>(x, vp, CIN, cb, 1, ctx);
    softmax_kernel<<<BATCH, 256, 0, stream>>>(ctx, sp, HW);

    // fused head -> alpha/beta, then elementwise epilogue
    fuse_kernel<<<BATCH, 512, 0, stream>>>(xm, w_vr, b_vr, w_up, b_up, ln_g, ln_b,
                                           w_red, bn_g, bn_b, bn_rm, bn_rv, w_sel,
                                           alpha, beta);
    final_kernel<<<(BATCH * CIN * HW) / 4 / 256, 256, 0, stream>>>(x, sp, alpha, beta, out);
}
